// GraphEncoder_85246510891752
// MI455X (gfx1250) — compile-verified, matrix operand order checked by reading, not run on hardware
//
#include <hip/hip_runtime.h>
#include <hip/hip_bf16.h>

typedef __attribute__((ext_vector_type(16))) __bf16 v16bf;
typedef __attribute__((ext_vector_type(4)))  __bf16 v4bf;
typedef __attribute__((ext_vector_type(8)))  float  v8f;
typedef int v4i_vs __attribute__((vector_size(16)));

#define IN_DIM  64
#define HID_DIM 128
#define OUT_DIM 32

#if __has_builtin(__builtin_amdgcn_global_load_async_to_lds_b128)
#define HAS_ASYNC_LDS 1
#else
#define HAS_ASYNC_LDS 0
#endif

// ---------------------------------------------------------------------------
// Async global->LDS staging (gfx1250 GLOBAL_LOAD_ASYNC_TO_LDS_B128, ASYNCcnt)
// ---------------------------------------------------------------------------
__device__ __forceinline__ void stage_b128(const float* __restrict__ gptr,
                                           float* __restrict__ lptr) {
#if HAS_ASYNC_LDS
    __builtin_amdgcn_global_load_async_to_lds_b128(
        (__attribute__((address_space(1))) v4i_vs*)gptr,
        (__attribute__((address_space(3))) v4i_vs*)lptr, 0, 0);
#else
    *(float4*)lptr = *(const float4*)gptr;
#endif
}

__device__ __forceinline__ void stage_wait() {
#if HAS_ASYNC_LDS
#if __has_builtin(__builtin_amdgcn_s_wait_asynccnt)
    __builtin_amdgcn_s_wait_asynccnt(0);
#else
    asm volatile("s_wait_asynccnt 0x0" ::: "memory");
#endif
#endif
}

// ---------------------------------------------------------------------------
// SELU
// ---------------------------------------------------------------------------
__device__ __forceinline__ float selu_f(float v) {
    const float alpha = 1.6732632423543772f;
    const float scale = 1.0507009873554805f;
    return v > 0.0f ? scale * v : scale * alpha * (__expf(v) - 1.0f);
}

// ---------------------------------------------------------------------------
// Degree / normalization kernels
// ---------------------------------------------------------------------------
__global__ __launch_bounds__(256) void init_deg_kernel(float* __restrict__ deg, int n) {
    int i = blockIdx.x * 256 + threadIdx.x;
    if (i < n) deg[i] = 1.0f;   // self loop
}

__global__ __launch_bounds__(256) void edge_deg_kernel(const int* __restrict__ dst,
                                                       float* __restrict__ deg, int e) {
    int i = blockIdx.x * 256 + threadIdx.x;
    if (i < e) unsafeAtomicAdd(deg + dst[i], 1.0f);
}

__global__ __launch_bounds__(256) void rsqrt_ip_kernel(float* __restrict__ d, int n) {
    int i = blockIdx.x * 256 + threadIdx.x;
    if (i < n) d[i] = rsqrtf(d[i]);   // deg >= 1 always (self loop)
}

// agg[i][c] = x[i][c] * dis[i]^2  (self-loop contribution; also zero-inits agg)
__global__ __launch_bounds__(256) void self_init_f32_kernel(const float* __restrict__ feat,
                                                            const float* __restrict__ dis,
                                                            float* __restrict__ agg, int n) {
    int idx = blockIdx.x * 256 + threadIdx.x;
    if (idx < (n << 6)) {
        int i = idx >> 6;
        float s = dis[i];
        agg[idx] = feat[idx] * s * s;
    }
}

// bf16-feature version (layer 2: h stored in bf16)
__global__ __launch_bounds__(256) void self_init_bf16_kernel(const __bf16* __restrict__ feat,
                                                             const float* __restrict__ dis,
                                                             float* __restrict__ agg, int n) {
    int idx = blockIdx.x * 256 + threadIdx.x;
    if (idx < (n << 7)) {
        int i = idx >> 7;
        float s = dis[i];
        agg[idx] = (float)feat[idx] * s * s;
    }
}

// ---------------------------------------------------------------------------
// Scatter kernels: one wave32 per edge
// ---------------------------------------------------------------------------
__global__ __launch_bounds__(256) void scatter64_kernel(const float* __restrict__ x,
                                                        const int* __restrict__ src,
                                                        const int* __restrict__ dst,
                                                        const float* __restrict__ dis,
                                                        float* __restrict__ agg, int e) {
    int edge = blockIdx.x * 8 + (threadIdx.x >> 5);
    int lane = threadIdx.x & 31;
    if (edge >= e) return;
    int s = src[edge], d = dst[edge];
    if (lane == 0 && edge + 8 < e)
        __builtin_prefetch(x + (size_t)src[edge + 8] * IN_DIM, 0, 0);
    float nrm = dis[s] * dis[d];
    float2 v = ((const float2*)(x + (size_t)s * IN_DIM))[lane];
    float* ar = agg + (size_t)d * IN_DIM + lane * 2;
    unsafeAtomicAdd(ar + 0, v.x * nrm);
    unsafeAtomicAdd(ar + 1, v.y * nrm);
}

// layer 2: gathers bf16 h rows (8 B/lane), accumulates f32 atomics
__global__ __launch_bounds__(256) void scatter128_kernel(const __bf16* __restrict__ h,
                                                         const int* __restrict__ src,
                                                         const int* __restrict__ dst,
                                                         const float* __restrict__ dis,
                                                         float* __restrict__ agg, int e) {
    int edge = blockIdx.x * 8 + (threadIdx.x >> 5);
    int lane = threadIdx.x & 31;
    if (edge >= e) return;
    int s = src[edge], d = dst[edge];
    if (lane == 0 && edge + 8 < e)
        __builtin_prefetch(h + (size_t)src[edge + 8] * HID_DIM, 0, 0);
    float nrm = dis[s] * dis[d];
    v4bf hv = *(const v4bf*)(h + (size_t)s * HID_DIM + lane * 4);
    float* ar = agg + (size_t)d * HID_DIM + lane * 4;
    unsafeAtomicAdd(ar + 0, (float)hv[0] * nrm);
    unsafeAtomicAdd(ar + 1, (float)hv[1] * nrm);
    unsafeAtomicAdd(ar + 2, (float)hv[2] * nrm);
    unsafeAtomicAdd(ar + 3, (float)hv[3] * nrm);
}

// ---------------------------------------------------------------------------
// WMMA fragment packing (ISA 7.12.2 bf16 16x32 A layout, mirrored for B)
//   A: lane = half*16 + r holds row M=r; VGPR v pair at
//      K = (v<4?0:16) + (half?8:0) + 2*(v&3) + {0,1}
// ---------------------------------------------------------------------------
__device__ __forceinline__ v16bf pack_A_bf16(const float* __restrict__ sA, int stride,
                                             int r, int half, int kOff) {
    v16bf a;
#pragma unroll
    for (int v = 0; v < 8; ++v) {
        int kb = ((v < 4) ? 0 : 16) + (half ? 8 : 0) + 2 * (v & 3) + kOff;
        a[2 * v]     = (__bf16)sA[r * stride + kb];
        a[2 * v + 1] = (__bf16)sA[r * stride + kb + 1];
    }
    return a;
}

__device__ __forceinline__ v16bf pack_B_bf16(const float* __restrict__ sW, int stride,
                                             int col, int half, int kOff) {
    v16bf b;
#pragma unroll
    for (int v = 0; v < 8; ++v) {
        int kb = ((v < 4) ? 0 : 16) + (half ? 8 : 0) + 2 * (v & 3) + kOff;
        b[2 * v]     = (__bf16)sW[(kb)     * stride + col];
        b[2 * v + 1] = (__bf16)sW[(kb + 1) * stride + col];
    }
    return b;
}

// ---------------------------------------------------------------------------
// GEMM1 + bias + SELU:  h[N,128](bf16) = selu(aggx[N,64] @ W1[64,128] + b1)
// Block: 256 thr = 8 waves. Each iteration covers 32 rows x 128 cols:
// wave w owns col tile w for both 16-row groups (B fragments reused).
// ---------------------------------------------------------------------------
__global__ __launch_bounds__(256) void gemm1_selu_kernel(const float* __restrict__ aggx,
                                                         const float* __restrict__ W1,
                                                         const float* __restrict__ b1,
                                                         __bf16* __restrict__ h, int n) {
    __shared__ float sW[64 * 132];
    __shared__ float sA[32 * 68];
    const int tid = threadIdx.x;
    for (int i = tid; i < 64 * 128; i += 256) {
        int k = i >> 7, c = i & 127;
        sW[k * 132 + c] = W1[i];
    }
    __syncthreads();

    const int wave = tid >> 5, lane = tid & 31;
    const int half = lane >> 4, nn = lane & 15;
    const int col = wave * 16 + nn;

    // loop-invariant: B fragments + bias
    v16bf bfrag0 = pack_B_bf16(sW, 132, col, half, 0);
    v16bf bfrag1 = pack_B_bf16(sW, 132, col, half, 32);
    const float bias = b1[col];

    const int numTiles = (n + 31) >> 5;
    for (int tile = blockIdx.x; tile < numTiles; tile += gridDim.x) {
        const int rowBase = tile << 5;
        __syncthreads();
        // stage A 32x64 via async global->LDS b128 (2 per thread)
#pragma unroll
        for (int j = 0; j < 2; ++j) {
            int idx = tid + j * 256;
            int r = idx >> 4;
            int c = (idx & 15) << 2;
            int gr = rowBase + r;
            float* lp = &sA[r * 68 + c];
            if (gr < n) {
                stage_b128(aggx + (size_t)gr * IN_DIM + c, lp);
            } else {
                lp[0] = lp[1] = lp[2] = lp[3] = 0.f;
            }
        }
        stage_wait();
        __syncthreads();

        v8f acc0 = {}, acc1 = {};
#pragma unroll
        for (int kc = 0; kc < 2; ++kc) {
            v16bf bf = kc ? bfrag1 : bfrag0;
            v16bf a0 = pack_A_bf16(sA, 68, nn, half, kc * 32);
            acc0 = __builtin_amdgcn_wmma_f32_16x16x32_bf16(false, a0, false, bf,
                                                           (short)0, acc0, false, false);
            v16bf a1 = pack_A_bf16(sA, 68, nn + 16, half, kc * 32);
            acc1 = __builtin_amdgcn_wmma_f32_16x16x32_bf16(false, a1, false, bf,
                                                           (short)0, acc1, false, false);
        }

        if (rowBase + 32 <= n) {   // full-tile fast path: unrolled, unguarded
#pragma unroll
            for (int v = 0; v < 8; ++v) {
                int r0 = rowBase + v + half * 8;
                h[(size_t)r0 * HID_DIM + col]        = (__bf16)selu_f(acc0[v] + bias);
                h[(size_t)(r0 + 16) * HID_DIM + col] = (__bf16)selu_f(acc1[v] + bias);
            }
        } else {
#pragma unroll
            for (int v = 0; v < 8; ++v) {
                int r0 = rowBase + v + half * 8;
                if (r0 < n)
                    h[(size_t)r0 * HID_DIM + col]        = (__bf16)selu_f(acc0[v] + bias);
                if (r0 + 16 < n)
                    h[(size_t)(r0 + 16) * HID_DIM + col] = (__bf16)selu_f(acc1[v] + bias);
            }
        }
    }
}

// ---------------------------------------------------------------------------
// GEMM2: mu = aggh @ W_mu + b_mu ; lv = aggh @ W_lv + b_lv  (both 128->32)
// Combined 64-wide weight tile in LDS. Block: 8 waves cover 32 rows x 64 cols.
// ---------------------------------------------------------------------------
__global__ __launch_bounds__(256) void gemm2_kernel(const float* __restrict__ aggh,
                                                    const float* __restrict__ Wmu,
                                                    const float* __restrict__ bmu,
                                                    const float* __restrict__ Wlv,
                                                    const float* __restrict__ blv,
                                                    float* __restrict__ out, int n) {
    __shared__ float sW[128 * 68];
    __shared__ float sA[32 * 132];
    const int tid = threadIdx.x;
    for (int i = tid; i < 128 * 64; i += 256) {
        int k = i >> 6, c = i & 63;
        sW[k * 68 + c] = (c < 32) ? Wmu[k * 32 + c] : Wlv[k * 32 + (c - 32)];
    }
    __syncthreads();

    const int wave = tid >> 5, lane = tid & 31;
    const int half = lane >> 4, nn = lane & 15;
    const int rg = wave >> 2;          // row group 0/1 (16 rows each)
    const int which = wave & 3;        // 0,1 -> mu cols; 2,3 -> lv cols
    const int cc = which * 16 + nn;    // combined column 0..63
    const int mat = cc >> 5;           // 0=mu, 1=lv
    const int oc = cc & 31;            // output column

    v16bf bfrag[4];
#pragma unroll
    for (int kc = 0; kc < 4; ++kc)
        bfrag[kc] = pack_B_bf16(sW, 68, cc, half, kc * 32);

    const float bias = (mat ? blv : bmu)[oc];
    float* obase = out + (size_t)mat * (size_t)n * OUT_DIM;

    const int numTiles = (n + 31) >> 5;
    for (int tile = blockIdx.x; tile < numTiles; tile += gridDim.x) {
        const int rowBase = tile << 5;
        __syncthreads();
        // stage A 32x128 via async global->LDS b128 (4 per thread)
#pragma unroll
        for (int j = 0; j < 4; ++j) {
            int idx = tid + j * 256;
            int r = idx >> 5;
            int c = (idx & 31) << 2;
            int gr = rowBase + r;
            float* lp = &sA[r * 132 + c];
            if (gr < n) {
                stage_b128(aggh + (size_t)gr * HID_DIM + c, lp);
            } else {
                lp[0] = lp[1] = lp[2] = lp[3] = 0.f;
            }
        }
        stage_wait();
        __syncthreads();

        v8f acc = {};
#pragma unroll
        for (int kc = 0; kc < 4; ++kc) {
            v16bf a = pack_A_bf16(sA, 132, rg * 16 + nn, half, kc * 32);
            acc = __builtin_amdgcn_wmma_f32_16x16x32_bf16(false, a, false, bfrag[kc],
                                                          (short)0, acc, false, false);
        }

        const int rBase = rowBase + rg * 16;
        if (rBase + 16 <= n) {   // full-tile fast path
#pragma unroll
            for (int v = 0; v < 8; ++v) {
                int row = rBase + v + half * 8;
                obase[(size_t)row * OUT_DIM + oc] = acc[v] + bias;
            }
        } else {
#pragma unroll
            for (int v = 0; v < 8; ++v) {
                int row = rBase + v + half * 8;
                if (row < n) obase[(size_t)row * OUT_DIM + oc] = acc[v] + bias;
            }
        }
    }
}

// ---------------------------------------------------------------------------
// Launch
// ---------------------------------------------------------------------------
extern "C" void kernel_launch(void* const* d_in, const int* in_sizes, int n_in,
                              void* d_out, int out_size, void* d_ws, size_t ws_size,
                              hipStream_t stream) {
    const float* x   = (const float*)d_in[0];
    const int*   ei  = (const int*)  d_in[1];
    const float* W1  = (const float*)d_in[2];
    const float* b1  = (const float*)d_in[3];
    const float* Wmu = (const float*)d_in[4];
    const float* bmu = (const float*)d_in[5];
    const float* Wlv = (const float*)d_in[6];
    const float* blv = (const float*)d_in[7];
    float* out = (float*)d_out;

    const int N = in_sizes[0] / IN_DIM;
    const int E = in_sizes[1] / 2;
    const int* src = ei;
    const int* dst = ei + E;

    // workspace: dis[N] f32 | h[N*128] bf16 | shared region (agg_x / agg_h) f32
    float*  dis = (float*)d_ws;
    __bf16* h   = (__bf16*)(dis + (size_t)N);
    float*  agg = (float*)(h + (size_t)N * HID_DIM);

    const int tB = 256;
    const int gN   = (N + tB - 1) / tB;
    const int gE   = (E + tB - 1) / tB;
    const int gEdg = (E + 7) / 8;           // 8 edges (waves) per block

    // 1) degrees (reuse dis buffer), 2) dis = rsqrt(deg)
    init_deg_kernel<<<gN, tB, 0, stream>>>(dis, N);
    edge_deg_kernel<<<gE, tB, 0, stream>>>(dst, dis, E);
    rsqrt_ip_kernel<<<gN, tB, 0, stream>>>(dis, N);

    // 3) layer-1 aggregation in input space (64 ch): self loops + edge scatter
    self_init_f32_kernel<<<(N * IN_DIM + tB - 1) / tB, tB, 0, stream>>>(x, dis, agg, N);
    scatter64_kernel<<<gEdg, tB, 0, stream>>>(x, src, dst, dis, agg, E);

    // 4) h(bf16) = selu(agg @ W1 + b1)  (WMMA bf16)
    int t1 = (N + 31) / 32;
    gemm1_selu_kernel<<<(t1 < 4096 ? t1 : 4096), tB, 0, stream>>>(agg, W1, b1, h, N);

    // 5) layer-2 aggregation of h (128 ch), shared by both heads
    self_init_bf16_kernel<<<(N * HID_DIM + tB - 1) / tB, tB, 0, stream>>>(h, dis, agg, N);
    scatter128_kernel<<<gEdg, tB, 0, stream>>>(h, src, dst, dis, agg, E);

    // 6) mu / logvar heads (WMMA bf16)
    int t2 = (N + 31) / 32;
    gemm2_kernel<<<(t2 < 4096 ? t2 : 4096), tB, 0, stream>>>(agg, Wmu, bmu, Wlv, blv, out, N);
}
